// ChsyConv_17678085390990
// MI455X (gfx1250) — matrix operand order repaired and checked by reading, not compiled, not run on hardware
//
#include <hip/hip_runtime.h>
#include <hip/hip_bf16.h>
#include <math.h>

// Problem constants (reference: B=16, N=8192, D=256, K=16)
#define B_    16
#define N_    8192
#define D_    256
#define KORD  16
#define LOG2N 13

#define MGRP  4      // row-groups of 16 per block -> M-tile = 64

typedef __attribute__((ext_vector_type(16))) __bf16 v16bf;
typedef __attribute__((ext_vector_type(8)))  float  v8f;

static __device__ __forceinline__ unsigned short f32_to_bf16_rne(float f) {
    union { float f; unsigned int u; } v; v.f = f;
    unsigned int u = v.u;
    if ((u & 0x7F800000u) == 0x7F800000u && (u & 0x007FFFFFu) != 0u)
        return (unsigned short)((u >> 16) | 0x0040u);     // quiet NaN
    unsigned int r = u + 0x7FFFu + ((u >> 16) & 1u);      // round-to-nearest-even
    return (unsigned short)(r >> 16);
}

static __device__ __forceinline__ float gelu_exact(float x) {
    return 0.5f * x * (1.0f + erff(x * 0.70710678118654752440f));
}

// ---------------------------------------------------------------------------
// K0: repack both weights (f32, row-major [K][256]) into WMMA B-fragment
// layout, bf16. Fragment (w, s, ct): lane c<16 holds col ct*16+c, K=32s+0..15;
// lane c+16 holds same col, K=32s+16..31 (contiguous k order, matching the A tiles).
// 256 blocks x 32 threads; frag linear index == blockIdx.x.
// ---------------------------------------------------------------------------
__global__ void pack_weights_kernel(const float* __restrict__ weig,
                                    const float* __restrict__ wval,
                                    unsigned short* __restrict__ bpack) {
    const int bid  = blockIdx.x;       // w*128 + s*16 + ct
    const int lane = threadIdx.x;      // 32 threads
    const int w  = bid >> 7;
    const int s  = (bid >> 4) & 7;
    const int ct = bid & 15;
    const float* W = w ? wval : weig;
    const int col = ct * 16 + (lane & 15);
    const int kb  = s * 32 + ((lane < 16) ? 0 : 16);
    unsigned short tmp[16];
#pragma unroll
    for (int e = 0; e < 16; ++e)
        tmp[e] = f32_to_bf16_rne(W[(size_t)(kb + e) * D_ + col]);
    unsigned int* dst = (unsigned int*)(bpack + ((size_t)bid * 32 + lane) * 16);
#pragma unroll
    for (int i = 0; i < 8; ++i)
        dst[i] = (unsigned int)tmp[2 * i] | ((unsigned int)tmp[2 * i + 1] << 16);
}

// ---------------------------------------------------------------------------
// K1: fused dual GEMM (bf16 WMMA, f32 accum) + exact GELU.
// One block = 64 rows of embed (4 row-groups of 16; 64 | 8192, so a block
// never crosses a batch). B fragments are loaded once per k-step and reused
// across the 4 row-groups (4x L2 traffic reduction vs a 16-row tile).
// Waves 0-3: eigen GEMM (cols 0..255) -> GELU -> row-sum -> pooled[B*N].
// Waves 4-7: value GEMM (cols 0..255) -> GELU -> valuews[B*N*D].
// ---------------------------------------------------------------------------
__global__ void __launch_bounds__(256) gemm_gelu_kernel(
    const float* __restrict__ embed,
    const unsigned short* __restrict__ bpack,
    float* __restrict__ valuews,
    float* __restrict__ pooledws) {

    __shared__ __attribute__((aligned(32))) unsigned short Atile[MGRP][8][32][16]; // 32 KB
    __shared__ float psum[MGRP][4][32][8];                                         // 16 KB

    const int tid = threadIdx.x;
    const size_t row0 = (size_t)blockIdx.x * (16 * MGRP);   // flat row b*N + n

    // Phase 1: cooperative f32->bf16 A tiles in A-fragment layout, contiguous k order.
    // lane<16: row=lane, elems 0..15 -> K=32s+0..15
    // lane>=16: row=lane-16, elems 0..15 -> K=32s+16..31
#pragma unroll
    for (int i = 0; i < MGRP * 16; ++i) {
        const int q    = tid + i * 256;     // 0..16383
        const int g    = q >> 12;
        const int qq   = q & 4095;
        const int s    = qq >> 9;
        const int lane = (qq >> 4) & 31;
        const int el   = qq & 15;
        const int row  = g * 16 + (lane & 15);
        const int kb   = s * 32 + ((lane < 16) ? 0 : 16);
        const int k    = kb + el;
        Atile[g][s][lane][el] = f32_to_bf16_rne(embed[(row0 + row) * D_ + k]);
    }
    __syncthreads();

    const int wave   = tid >> 5;
    const int lane   = tid & 31;
    const int branch = wave >> 2;        // 0 = eigen, 1 = value
    const int ct0    = (wave & 3) * 4;   // 4 column tiles of 16 per wave

    const v16bf* bp = (const v16bf*)bpack + (size_t)branch * (8 * 16 * 32);

    v8f acc[MGRP][4];
#pragma unroll
    for (int g = 0; g < MGRP; ++g)
#pragma unroll
        for (int c = 0; c < 4; ++c) acc[g][c] = v8f{};

    for (int s = 0; s < 8; ++s) {
        v16bf bfr[4];
#pragma unroll
        for (int c = 0; c < 4; ++c)
            bfr[c] = bp[(size_t)((s * 16 + ct0 + c) * 32) + lane];
#pragma unroll
        for (int g = 0; g < MGRP; ++g) {
            const v16bf a = *(const v16bf*)&Atile[g][s][lane][0];
#pragma unroll
            for (int c = 0; c < 4; ++c)
                acc[g][c] = __builtin_amdgcn_wmma_f32_16x16x32_bf16(
                    false, a, false, bfr[c], (short)0, acc[g][c], false, false);
        }
    }

    // C/D layout: VGPR r, lane<16 -> (M=r, N=lane); lane>=16 -> (M=r+8, N=lane-16)
    const int mh = (lane < 16) ? 0 : 8;
    const int cl = lane & 15;
    if (branch == 1) {
#pragma unroll
        for (int g = 0; g < MGRP; ++g)
#pragma unroll
            for (int c = 0; c < 4; ++c) {
                const int col = (ct0 + c) * 16 + cl;
#pragma unroll
                for (int r = 0; r < 8; ++r)
                    valuews[(row0 + g * 16 + r + mh) * D_ + col] =
                        gelu_exact(acc[g][c][r]);
            }
    } else {
#pragma unroll
        for (int g = 0; g < MGRP; ++g) {
            float ls[8];
#pragma unroll
            for (int r = 0; r < 8; ++r) ls[r] = 0.0f;
#pragma unroll
            for (int c = 0; c < 4; ++c)
#pragma unroll
                for (int r = 0; r < 8; ++r)
                    ls[r] += gelu_exact(acc[g][c][r]);
#pragma unroll
            for (int r = 0; r < 8; ++r) psum[g][wave][lane][r] = ls[r];
        }
    }
    __syncthreads();

    // Deterministic fixed-order reduction of pooled row sums (no float atomics).
    if (tid < 16 * MGRP) {
        const int g = tid >> 4;
        const int m = tid & 15;
        const int half = (m < 8) ? 0 : 16;
        const int r = m & 7;
        float sum = 0.0f;
        for (int w = 0; w < 4; ++w)
            for (int c = 0; c < 16; ++c)
                sum += psum[g][w][half + c][r];
        pooledws[row0 + g * 16 + m] = sum;
    }
}

// ---------------------------------------------------------------------------
// K2: per-batch min/max normalize + Chebyshev-Jackson recurrence -> cheb[B*N]
// ---------------------------------------------------------------------------
__global__ void __launch_bounds__(256) cheb_kernel(
    const float* __restrict__ pooledws,
    const float* __restrict__ gibbs,
    float* __restrict__ chebws) {
    const int b = blockIdx.x;
    const int tid = threadIdx.x;
    __shared__ float smin[256], smax[256], sd[KORD + 1];

    const float* p = pooledws + (size_t)b * N_;
    float mn = 3.402823466e38f, mx = -3.402823466e38f;
#pragma unroll
    for (int i = 0; i < N_ / 256; ++i) {
        const float v = p[tid + i * 256];
        mn = fminf(mn, v);
        mx = fmaxf(mx, v);
    }
    smin[tid] = mn; smax[tid] = mx;
    if (tid <= KORD) sd[tid] = gibbs[b * (KORD + 1) + tid];
    __syncthreads();
    for (int s = 128; s > 0; s >>= 1) {
        if (tid < s) {
            smin[tid] = fminf(smin[tid], smin[tid + s]);
            smax[tid] = fmaxf(smax[tid], smax[tid + s]);
        }
        __syncthreads();
    }
    mn = smin[0]; mx = smax[0];
    const float sc = 2.0f / (mx - mn);

    float* cw = chebws + (size_t)b * N_;
#pragma unroll
    for (int i = 0; i < N_ / 256; ++i) {
        const int n = tid + i * 256;
        const float e = (p[n] - mn) * sc - 1.0f;
        float t0 = 1.0f, t1 = e;
        float ch = t0 + t1 * sd[1];
#pragma unroll
        for (int k = 2; k <= KORD; ++k) {
            const float t2 = 2.0f * e * t1 - t0;
            ch += t2 * sd[k];
            t0 = t1; t1 = t2;
        }
        cw[n] = ch;
    }
}

// ---------------------------------------------------------------------------
// K3: per-(b,d) column: FFT_8192 -> * cheb[b,k] -> IFFT_8192, all in LDS.
// Forward: radix-2 DIT (bit-reversed in, natural out).
// Inverse: radix-2 DIF (natural in, bit-reversed out; fixed on store).
// Output complex64-interleaved at [b][n][d].
// ---------------------------------------------------------------------------
__global__ void __launch_bounds__(256) fft_conv_kernel(
    const float* __restrict__ valuews,
    const float* __restrict__ chebws,
    float2* __restrict__ out) {
    __shared__ float sre[N_];   // 32 KB
    __shared__ float sim[N_];   // 32 KB (64 KB total of the 320 KB/WGP)

    const int tid = threadIdx.x;
    const int bid = blockIdx.x;
    const int b = bid >> 8;     // D_ = 256
    const int d = bid & (D_ - 1);

    const float* src = valuews + (size_t)b * N_ * D_ + d;
    const float* cw  = chebws + (size_t)b * N_;

#pragma unroll
    for (int i = 0; i < N_ / 256; ++i) {
        const int n = tid + i * 256;
        const int r = (int)(__brev((unsigned)n) >> (32 - LOG2N));
        sre[r] = src[(size_t)n * D_];
        sim[r] = 0.0f;
        __builtin_prefetch(&cw[n], 0, 0);   // global_prefetch_b8 for cheb row
    }
    __syncthreads();

    // Forward FFT (DIT)
    for (int st = 1; st <= LOG2N; ++st) {
        const int half = 1 << (st - 1);
        const float ang = -6.28318530717958647692f / (float)(1 << st);
        for (int t = tid; t < N_ / 2; t += 256) {
            const int j  = t & (half - 1);
            const int i0 = ((t >> (st - 1)) << st) + j;
            const int i1 = i0 + half;
            float c, sn;
            __sincosf(ang * (float)j, &sn, &c);
            const float ar = sre[i0], ai = sim[i0];
            const float br = sre[i1], bi = sim[i1];
            const float tr = br * c - bi * sn;
            const float ti = br * sn + bi * c;
            sre[i0] = ar + tr; sim[i0] = ai + ti;
            sre[i1] = ar - tr; sim[i1] = ai - ti;
        }
        __syncthreads();
    }

    // Frequency-domain scale by cheb (real)
#pragma unroll
    for (int i = 0; i < N_ / 256; ++i) {
        const int k = tid + i * 256;
        const float f = cw[k];
        sre[k] *= f;
        sim[k] *= f;
    }
    __syncthreads();

    // Inverse FFT (DIF, conjugate twiddles)
    for (int st = LOG2N; st >= 1; --st) {
        const int half = 1 << (st - 1);
        const float ang = 6.28318530717958647692f / (float)(1 << st);
        for (int t = tid; t < N_ / 2; t += 256) {
            const int j  = t & (half - 1);
            const int i0 = ((t >> (st - 1)) << st) + j;
            const int i1 = i0 + half;
            float c, sn;
            __sincosf(ang * (float)j, &sn, &c);
            const float ar = sre[i0], ai = sim[i0];
            const float br = sre[i1], bi = sim[i1];
            sre[i0] = ar + br; sim[i0] = ai + bi;
            const float dr = ar - br, di = ai - bi;
            sre[i1] = dr * c - di * sn;
            sim[i1] = dr * sn + di * c;
        }
        __syncthreads();
    }

    const float inv = 1.0f / (float)N_;
#pragma unroll
    for (int i = 0; i < N_ / 256; ++i) {
        const int n = tid + i * 256;
        const int r = (int)(__brev((unsigned)n) >> (32 - LOG2N));
        out[((size_t)b * N_ + n) * D_ + d] = make_float2(sre[r] * inv, sim[r] * inv);
    }
}

// ---------------------------------------------------------------------------
// Workspace layout (floats): value[B*N*D] | pooled[B*N] | cheb[B*N] | bpack(bf16)
// total ~= 134.2 MB + 0.5 MB + 0.5 MB + 0.25 MB
// ---------------------------------------------------------------------------
extern "C" void kernel_launch(void* const* d_in, const int* in_sizes, int n_in,
                              void* d_out, int out_size, void* d_ws, size_t ws_size,
                              hipStream_t stream) {
    (void)in_sizes; (void)n_in; (void)out_size; (void)ws_size;
    const float* embed = (const float*)d_in[0];
    const float* weig  = (const float*)d_in[1];
    const float* wval  = (const float*)d_in[2];
    const float* gibbs = (const float*)d_in[3];

    float* valuews  = (float*)d_ws;
    float* pooledws = valuews + (size_t)B_ * N_ * D_;
    float* chebws   = pooledws + (size_t)B_ * N_;
    unsigned short* bpack = (unsigned short*)(chebws + (size_t)B_ * N_);

    pack_weights_kernel<<<256, 32, 0, stream>>>(weig, wval, bpack);
    gemm_gelu_kernel<<<(B_ * N_) / (16 * MGRP), 256, 0, stream>>>(embed, bpack, valuews, pooledws);
    cheb_kernel<<<B_, 256, 0, stream>>>(pooledws, gibbs, chebws);
    fft_conv_kernel<<<B_ * D_, 256, 0, stream>>>(valuews, chebws, (float2*)d_out);
}
